// DecoderRNN_7035156430962
// MI455X (gfx1250) — matrix operand order repaired and checked by reading, not compiled, not run on hardware
//
#include <hip/hip_runtime.h>
#include <hip/hip_bf16.h>
#include <math.h>

typedef __attribute__((ext_vector_type(16))) __bf16 v16bf;
typedef __attribute__((ext_vector_type(8)))  __bf16 bf16x8;
typedef __attribute__((ext_vector_type(8)))  float  v8f;

namespace {
constexpr int NB   = 64;          // batch
constexpr int EMB  = 512;         // E
constexpr int HE2  = 1024;        // 2*H_ENC (context width)
constexpr int HID  = 1024;        // H
constexpr int G3   = 3 * HID;     // 3072
constexpr int KIH  = EMB + HE2;   // 1536 (W_ih inner dim)
constexpr int VOC  = 32000;
constexpr int TSTEPS = 64;
constexpr int SOS_TOK = 1;
constexpr int EOS_TOK = 2;

constexpr int NT_LOG = 4;                    // N-tiles per wave (logits)
constexpr int NSLAB  = VOC / (16 * NT_LOG);  // 500 partial slabs

// ---- workspace layout (bytes) ----
constexpr size_t O_WIH  = 0;
constexpr size_t SZ_WIH = (size_t)G3 * KIH * 2;
constexpr size_t O_WHH  = O_WIH + SZ_WIH;
constexpr size_t SZ_WHH = (size_t)G3 * HID * 2;
constexpr size_t O_WOUT = O_WHH + SZ_WHH;
constexpr size_t SZ_WOUT= (size_t)VOC * HID * 2;
constexpr size_t O_WHW  = O_WOUT + SZ_WOUT;
constexpr size_t SZ_WHW = (size_t)HID * HE2 * 2;
constexpr size_t O_FLAT = O_WHW + SZ_WHW;
constexpr size_t SZ_FLAT= (size_t)NB * HE2 * 2;
constexpr size_t O_CTX  = O_FLAT + SZ_FLAT;
constexpr size_t SZ_CTX = (size_t)NB * G3 * 4;
constexpr size_t O_G1   = O_CTX + SZ_CTX;
constexpr size_t O_G2   = O_G1 + SZ_CTX;
constexpr size_t O_H    = O_G2 + SZ_CTX;
constexpr size_t SZ_H   = (size_t)NB * HID * 4;
constexpr size_t O_HB   = O_H + SZ_H;
constexpr size_t SZ_HB  = (size_t)NB * HID * 2;
constexpr size_t O_XB   = O_HB + SZ_HB;
constexpr size_t SZ_XB  = (size_t)NB * EMB * 2;
constexpr size_t O_PV   = O_XB + SZ_XB;
constexpr size_t SZ_PV  = (size_t)NB * NSLAB * 4;
constexpr size_t O_PI   = O_PV + SZ_PV;
constexpr size_t O_INP  = O_PI + SZ_PV;
constexpr size_t O_FIN  = O_INP + 256;
} // namespace

// ---------------- WMMA fragment helpers (wave32, 16x16x32 bf16) ----------------
// A (16x32, row-major, lda elems): lane m = lane&15; ko = 8*(lane>>4);
//   VGPR0-3 = K[ko..ko+7], VGPR4-7 = K[ko+16..ko+23]
__device__ __forceinline__ v16bf load_a_frag(const __bf16* A, int lda, int row0,
                                             int k0, int lane) {
  int m  = row0 + (lane & 15);
  int ko = k0 + ((lane >> 4) << 3);
  const bf16x8* p = reinterpret_cast<const bf16x8*>(A + (size_t)m * lda + ko);
  bf16x8 lo = p[0];
  bf16x8 hi = p[2];  // +16 elements
  return __builtin_shufflevector(lo, hi, 0,1,2,3,4,5,6,7,8,9,10,11,12,13,14,15);
}

// B (32x16 KxN) sourced from row-major W[N,K] (W row = B column):
//   lane n = lane&15; ko = 8*(lane>>4); VGPR0-3 = K[ko..ko+7], VGPR4-7 = K[ko+16..ko+23]
__device__ __forceinline__ v16bf load_b_frag(const __bf16* W, int ldw, int n0,
                                             int k0, int lane) {
  int n  = n0 + (lane & 15);
  int kb = k0 + ((lane >> 4) << 3);
  const bf16x8* p = reinterpret_cast<const bf16x8*>(W + (size_t)n * ldw + kb);
  bf16x8 lo = p[0];
  bf16x8 hi = p[2];  // +16 elements
  return __builtin_shufflevector(lo, hi, 0,1,2,3,4,5,6,7,8,9,10,11,12,13,14,15);
}

// 64x16 slab: out += A[64,K] * W[n0..n0+15, K]^T
__device__ __forceinline__ void wmma_slab(const __bf16* A, int lda,
                                          const __bf16* W, int ldw,
                                          int n0, int K, v8f acc[4], int lane) {
  for (int k0 = 0; k0 < K; k0 += 32) {
    v16bf b = load_b_frag(W, ldw, n0, k0, lane);
#pragma unroll
    for (int mt = 0; mt < 4; ++mt) {
      v16bf a = load_a_frag(A, lda, mt * 16, k0, lane);
      acc[mt] = __builtin_amdgcn_wmma_f32_16x16x32_bf16(
          false, a, false, b, (short)0, acc[mt], false, false);
    }
  }
}

// D element (m,n): VGPR v = m%8, lane = n + 16*(m/8)
__device__ __forceinline__ void store_slab_f32(float* C, int ldc, int col0,
                                               const v8f acc[4], int lane) {
  int n  = col0 + (lane & 15);
  int hb = (lane >> 4) << 3;
#pragma unroll
  for (int mt = 0; mt < 4; ++mt)
#pragma unroll
    for (int v = 0; v < 8; ++v)
      C[(size_t)(mt * 16 + hb + v) * ldc + n] = acc[mt][v];
}

// ---------------- kernels ----------------
__global__ void cvt_f32_bf16(const float* __restrict__ src, __bf16* __restrict__ dst, int n) {
  int i = blockIdx.x * blockDim.x + threadIdx.x;
  if (i < n) dst[i] = (__bf16)src[i];
}

// flat[b, i*512+j] = enc[i, b, j]; also init inp/finished
__global__ void build_flat(const float* __restrict__ enc, __bf16* __restrict__ flatbf,
                           int* inp, int* fin) {
  int idx = blockIdx.x * blockDim.x + threadIdx.x;
  if (idx < NB * HE2) {
    int b = idx >> 10, d = idx & 1023;
    int i = d >> 9,  j = d & 511;
    flatbf[idx] = (__bf16)enc[(size_t)(i * NB + b) * 512 + j];
  }
  if (idx < NB) { inp[idx] = SOS_TOK; fin[idx] = 0; }
}

// h0 = flat @ Wh_w^T + Wh_b   (64x1024, K=1024); also emit bf16 copy of h
__global__ void h0_kernel(const __bf16* flatbf, const __bf16* WhwB, const float* Whb,
                          float* hF, __bf16* hB) {
  int wid  = blockIdx.x * (blockDim.x >> 5) + (threadIdx.x >> 5);
  int lane = threadIdx.x & 31;
  if (wid >= HID / 16) return;
  int col0 = wid * 16;
  float bias = Whb[col0 + (lane & 15)];
  v8f acc[4];
#pragma unroll
  for (int mt = 0; mt < 4; ++mt)
#pragma unroll
    for (int v = 0; v < 8; ++v) acc[mt][v] = bias;
  wmma_slab(flatbf, HE2, WhwB, HE2, col0, HE2, acc, lane);
  store_slab_f32(hF, HID, col0, acc, lane);
  int n = col0 + (lane & 15), hb = (lane >> 4) << 3;
#pragma unroll
  for (int mt = 0; mt < 4; ++mt)
#pragma unroll
    for (int v = 0; v < 8; ++v)
      hB[(size_t)(mt * 16 + hb + v) * HID + n] = (__bf16)acc[mt][v];
}

// ctx_gi = context @ W_ih[:,512:]^T + b_ih   (64x3072, K=1024)
__global__ void ctx_kernel(const __bf16* flatbf, const __bf16* wihB, const float* bih,
                           float* ctxgi) {
  int wid  = blockIdx.x * (blockDim.x >> 5) + (threadIdx.x >> 5);
  int lane = threadIdx.x & 31;
  if (wid >= G3 / 16) return;
  int col0 = wid * 16;
  float bias = bih[col0 + (lane & 15)];
  v8f acc[4];
#pragma unroll
  for (int mt = 0; mt < 4; ++mt)
#pragma unroll
    for (int v = 0; v < 8; ++v) acc[mt][v] = bias;
  wmma_slab(flatbf, HE2, wihB + EMB, KIH, col0, HE2, acc, lane);
  store_slab_f32(ctxgi, G3, col0, acc, lane);
}

// x = bf16(embedding[inp[b]])
__global__ void embed_kernel(const float* __restrict__ emb, const int* __restrict__ inp,
                             __bf16* __restrict__ xbf) {
  int i = blockIdx.x * blockDim.x + threadIdx.x;
  if (i < NB * EMB) {
    int b = i >> 9, e = i & 511;
    xbf[i] = (__bf16)emb[(size_t)inp[b] * EMB + e];
  }
}

// G1 = x @ Wx^T + ctx_gi ; G2 = h @ W_hh^T + b_hh   (each 64x3072)
__global__ void gates_kernel(const __bf16* xbf, const __bf16* hB,
                             const __bf16* wihB, const __bf16* whhB,
                             const float* ctxgi, const float* bhh,
                             float* G1, float* G2) {
  int wid  = blockIdx.x * (blockDim.x >> 5) + (threadIdx.x >> 5);
  int lane = threadIdx.x & 31;
  if (wid >= 2 * (G3 / 16)) return;
  v8f acc[4];
  if (wid < G3 / 16) {
    int col0 = wid * 16;
    int n = col0 + (lane & 15), hb = (lane >> 4) << 3;
#pragma unroll
    for (int mt = 0; mt < 4; ++mt)
#pragma unroll
      for (int v = 0; v < 8; ++v)
        acc[mt][v] = ctxgi[(size_t)(mt * 16 + hb + v) * G3 + n];
    wmma_slab(xbf, EMB, wihB, KIH, col0, EMB, acc, lane);
    store_slab_f32(G1, G3, col0, acc, lane);
  } else {
    int col0 = (wid - G3 / 16) * 16;
    float bias = bhh[col0 + (lane & 15)];
#pragma unroll
    for (int mt = 0; mt < 4; ++mt)
#pragma unroll
      for (int v = 0; v < 8; ++v) acc[mt][v] = bias;
    wmma_slab(hB, HID, whhB, HID, col0, HID, acc, lane);
    store_slab_f32(G2, G3, col0, acc, lane);
  }
}

// GRU gate math -> new h (fp32 + bf16)
__global__ void gate_elem(const float* __restrict__ G1, const float* __restrict__ G2,
                          float* __restrict__ hF, __bf16* __restrict__ hB) {
  int i = blockIdx.x * blockDim.x + threadIdx.x;
  if (i >= NB * HID) return;
  int b = i >> 10, j = i & 1023;
  size_t base = (size_t)b * G3 + j;
  float r = 1.0f / (1.0f + expf(-(G1[base] + G2[base])));
  float z = 1.0f / (1.0f + expf(-(G1[base + HID] + G2[base + HID])));
  float n = tanhf(G1[base + 2 * HID] + r * G2[base + 2 * HID]);
  float hn = (1.0f - z) * n + z * hF[i];
  hF[i] = hn;
  hB[i] = (__bf16)hn;
}

// logits slab (64 x 64 cols) + per-row partial argmax
__global__ void logits_kernel(const __bf16* hB, const __bf16* Wo, const float* outb,
                              float* pval, int* pidx) {
  int wid  = blockIdx.x * (blockDim.x >> 5) + (threadIdx.x >> 5);
  int lane = threadIdx.x & 31;
  if (wid >= NSLAB) return;
  int colbase = wid * (16 * NT_LOG);
  v8f acc[NT_LOG][4];
#pragma unroll
  for (int nt = 0; nt < NT_LOG; ++nt)
#pragma unroll
    for (int mt = 0; mt < 4; ++mt) acc[nt][mt] = v8f{};
  for (int k0 = 0; k0 < HID; k0 += 32) {
    v16bf a[4];
#pragma unroll
    for (int mt = 0; mt < 4; ++mt) a[mt] = load_a_frag(hB, HID, mt * 16, k0, lane);
#pragma unroll
    for (int nt = 0; nt < NT_LOG; ++nt) {
      v16bf b = load_b_frag(Wo, HID, colbase + nt * 16, k0, lane);
#pragma unroll
      for (int mt = 0; mt < 4; ++mt)
        acc[nt][mt] = __builtin_amdgcn_wmma_f32_16x16x32_bf16(
            false, a[mt], false, b, (short)0, acc[nt][mt], false, false);
    }
  }
  int hi = lane >> 4, nl = lane & 15;
#pragma unroll
  for (int mt = 0; mt < 4; ++mt) {
#pragma unroll
    for (int v = 0; v < 8; ++v) {
      float bv = -3.4e38f; int bi = 0;
#pragma unroll
      for (int nt = 0; nt < NT_LOG; ++nt) {
        int col = colbase + nt * 16 + nl;
        float val = acc[nt][mt][v] + outb[col];
        if (val > bv) { bv = val; bi = col; }   // ascending col keeps first max
      }
      for (int off = 1; off < 16; off <<= 1) {  // reduce within 16-lane N group
        float ov = __shfl_xor(bv, off, 32);
        int   oi = __shfl_xor(bi, off, 32);
        if (ov > bv || (ov == bv && oi < bi)) { bv = ov; bi = oi; }
      }
      if (nl == 0) {
        int row = mt * 16 + hi * 8 + v;
        pval[(size_t)row * NSLAB + wid] = bv;
        pidx[(size_t)row * NSLAB + wid] = bi;
      }
    }
  }
}

// final argmax over slabs; update token feedback state; record token
__global__ void argmax_step(const float* __restrict__ pval, const int* __restrict__ pidx,
                            int* inp, int* fin, float* outTok, int t) {
  int b = threadIdx.x;
  if (b >= NB) return;
  float bv = -3.4e38f; int bi = 0;
  for (int s = 0; s < NSLAB; ++s) {
    float v = pval[(size_t)b * NSLAB + s];
    int   i = pidx[(size_t)b * NSLAB + s];
    if (v > bv || (v == bv && i < bi)) { bv = v; bi = i; }
  }
  int tok = bi;
  int f = fin[b] | (tok == EOS_TOK ? 1 : 0);
  fin[b] = f;
  inp[b] = f ? EOS_TOK : tok;
  outTok[(size_t)b * TSTEPS + t] = (float)tok;
}

__global__ void copy_h(const float* __restrict__ hF, float* __restrict__ dst) {
  int i = blockIdx.x * blockDim.x + threadIdx.x;
  if (i < NB * HID) dst[i] = hF[i];
}

// ---------------- launch ----------------
extern "C" void kernel_launch(void* const* d_in, const int* in_sizes, int n_in,
                              void* d_out, int out_size, void* d_ws, size_t ws_size,
                              hipStream_t stream) {
  const float* enc  = (const float*)d_in[0];
  const float* emb  = (const float*)d_in[1];
  const float* Whw  = (const float*)d_in[2];
  const float* Whb  = (const float*)d_in[3];
  const float* Wih  = (const float*)d_in[4];
  const float* Whh  = (const float*)d_in[5];
  const float* bih  = (const float*)d_in[6];
  const float* bhh  = (const float*)d_in[7];
  const float* outw = (const float*)d_in[8];
  const float* outb = (const float*)d_in[9];
  float* out = (float*)d_out;

  char* ws = (char*)d_ws;
  __bf16* wihB  = (__bf16*)(ws + O_WIH);
  __bf16* whhB  = (__bf16*)(ws + O_WHH);
  __bf16* woutB = (__bf16*)(ws + O_WOUT);
  __bf16* whwB  = (__bf16*)(ws + O_WHW);
  __bf16* flatB = (__bf16*)(ws + O_FLAT);
  float*  ctxgi = (float*)(ws + O_CTX);
  float*  G1    = (float*)(ws + O_G1);
  float*  G2    = (float*)(ws + O_G2);
  float*  hF    = (float*)(ws + O_H);
  __bf16* hB    = (__bf16*)(ws + O_HB);
  __bf16* xB    = (__bf16*)(ws + O_XB);
  float*  pval  = (float*)(ws + O_PV);
  int*    pidx  = (int*)(ws + O_PI);
  int*    inp   = (int*)(ws + O_INP);
  int*    fin   = (int*)(ws + O_FIN);

  const int TPB = 256;
  auto blocks = [](int n, int tpb) { return (n + tpb - 1) / tpb; };

  // one-time (per launch) weight conversion fp32 -> bf16
  cvt_f32_bf16<<<blocks(G3 * KIH, TPB), TPB, 0, stream>>>(Wih,  wihB,  G3 * KIH);
  cvt_f32_bf16<<<blocks(G3 * HID, TPB), TPB, 0, stream>>>(Whh,  whhB,  G3 * HID);
  cvt_f32_bf16<<<blocks(VOC * HID, TPB), TPB, 0, stream>>>(outw, woutB, VOC * HID);
  cvt_f32_bf16<<<blocks(HID * HE2, TPB), TPB, 0, stream>>>(Whw,  whwB,  HID * HE2);

  build_flat<<<blocks(NB * HE2, TPB), TPB, 0, stream>>>(enc, flatB, inp, fin);
  h0_kernel<<<blocks((HID / 16) * 32, TPB), TPB, 0, stream>>>(flatB, whwB, Whb, hF, hB);
  ctx_kernel<<<blocks((G3 / 16) * 32, TPB), TPB, 0, stream>>>(flatB, wihB, bih, ctxgi);

  for (int t = 0; t < TSTEPS; ++t) {
    embed_kernel<<<blocks(NB * EMB, TPB), TPB, 0, stream>>>(emb, inp, xB);
    gates_kernel<<<blocks(2 * (G3 / 16) * 32, TPB), TPB, 0, stream>>>(
        xB, hB, wihB, whhB, ctxgi, bhh, G1, G2);
    gate_elem<<<blocks(NB * HID, TPB), TPB, 0, stream>>>(G1, G2, hF, hB);
    logits_kernel<<<blocks(NSLAB * 32, TPB), TPB, 0, stream>>>(hB, woutB, outb, pval, pidx);
    argmax_step<<<1, NB, 0, stream>>>(pval, pidx, inp, fin, out, t);
  }
  copy_h<<<blocks(NB * HID, TPB), TPB, 0, stream>>>(hF, out + NB * TSTEPS);
}
